// DeformableConvNet_72645076844866
// MI455X (gfx1250) — matrix operand order repaired and checked by reading, not compiled, not run on hardware
//
#include <hip/hip_runtime.h>
#include <hip/hip_bf16.h>
#include <stdint.h>

// ---------------------------------------------------------------------------
// Deformable conv for MI455X (gfx1250, wave32, WMMA, async-to-LDS).
//   B=4, C=256, O=256, H=W=64, K=3, PAD=1
// Pipeline:
//   1) k_wcvt:    w_def f32 -> f16, k-interleaved Whk[k/32][m][k%32]
//   2) k_offconv: offsets = conv(x, w_off) + b_off       (f32, ws)
//   3) k_sample:  deformable bilinear sampling -> S f16, S[k/32][n][k%32]
//   4) k_gemm:    out = W * S + b_def  via v_wmma_f32_16x16x32_f16,
//                 128x128 block tile, 64x32 wave tiles (8 WMMA/wave/iter),
//                 double-buffered global_load_async_to_lds_b128 staging
// ---------------------------------------------------------------------------

typedef __attribute__((ext_vector_type(16))) _Float16 v16h;
typedef __attribute__((ext_vector_type(8)))  float    v8f;

#define BB   4
#define CC   256
#define OO   256
#define HH   64
#define WW   64
#define HW   4096          // 64*64
#define CK   2304          // C*9
#define NN   16384         // B*HW
#define NKT  72            // CK/32 k-tiles

#define LDROW 40           // 32 halves + 8 pad (80B row: 16B-aligned, conflict-free)

union F16x16 { v16h v; _Float16 h[16]; };
union F32x8  { v8f  v; float    f[8];  };

// async copy global -> LDS, 16 bytes per lane (ASYNCcnt-tracked)
__device__ __forceinline__ void async_b128(uint32_t lds_byte_off, const void* gptr) {
  asm volatile("global_load_async_to_lds_b128 %0, %1, off"
               :: "v"(lds_byte_off), "v"((unsigned long long)(uintptr_t)gptr)
               : "memory");
}
__device__ __forceinline__ void wait_async0() {
  asm volatile("s_wait_asynccnt 0" ::: "memory");
}

// ---------------------------------------------------------------------------
// 1) weight convert + swizzle: w_def (O,C,3,3) f32 -> Whk[k/32][256][32] f16
// ---------------------------------------------------------------------------
__global__ __launch_bounds__(256) void k_wcvt(const float* __restrict__ wd,
                                              _Float16* __restrict__ whk) {
  int tid = blockIdx.x * 256 + threadIdx.x;   // exactly O*CK = 589824 threads
  int o = tid / CK;
  int k = tid % CK;
  whk[(size_t)(k >> 5) * (OO * 32) + (o << 5) + (k & 31)] = (_Float16)wd[tid];
}

// ---------------------------------------------------------------------------
// 2) offset-predictor conv (direct, VALU; ~1.4 GFLOP).
// ---------------------------------------------------------------------------
__global__ __launch_bounds__(256) void k_offconv(const float* __restrict__ x,
                                                 const float* __restrict__ w_off,
                                                 const float* __restrict__ b_off,
                                                 float* __restrict__ offs) {
  int tid = blockIdx.x * 256 + threadIdx.x;   // B*18*HW = 294912 threads
  int hw  = tid & (HW - 1);
  int oc  = (tid >> 12) % 18;
  int b   = tid / (18 * HW);
  int h   = hw >> 6;
  int w   = hw & 63;

  float acc = b_off[oc];
  const float* xb = x + ((size_t)b << 20);
  const float* wb = w_off + oc * CK;
  for (int c = 0; c < CC; ++c) {
    const float* xc = xb + (c << 12);
    const float* wc = wb + c * 9;
#pragma unroll
    for (int t = 0; t < 9; ++t) {
      int yy = h + t / 3 - 1;
      int xx = w + t % 3 - 1;
      if (yy >= 0 && yy < HH && xx >= 0 && xx < WW)
        acc += xc[yy * WW + xx] * wc[t];
    }
  }
  offs[(b * 18 + oc) * HW + hw] = acc;
}

// ---------------------------------------------------------------------------
// 3) deformable bilinear sampling -> S[k/32][n][k%32] f16
//    One thread per (b,c,hw); its 9 taps land at consecutive k slots.
// ---------------------------------------------------------------------------
__global__ __launch_bounds__(256) void k_sample(const float* __restrict__ x,
                                                const float* __restrict__ offs,
                                                _Float16* __restrict__ S) {
  int tid = blockIdx.x * 256 + threadIdx.x;   // B*C*HW = 4194304 threads
  int hw  = tid & (HW - 1);
  int c   = (tid >> 12) & (CC - 1);
  int b   = tid >> 20;
  int h   = hw >> 6;
  int w   = hw & 63;

  const float* xc = x + (((size_t)(b * CC + c)) << 12);
  int offbase = (b * 18) * HW + hw;
  const size_t n32 = (size_t)((b << 12) + hw) << 5;   // n*32
  const int kbase = c * 9;

#pragma unroll
  for (int kk = 0; kk < 9; ++kk) {
    float dy = offs[offbase + (2 * kk) * HW];
    float dx = offs[offbase + (2 * kk + 1) * HW];
    float py = dy + (float)(h + kk / 3 - 1);
    float px = dx + (float)(w + kk % 3 - 1);
    float y0f = floorf(py);
    float x0f = floorf(px);
    int   y0  = (int)y0f;
    int   x0  = (int)x0f;
    float ly  = py - y0f;
    float lx  = px - x0f;
    float v   = 0.f;
#pragma unroll
    for (int dyi = 0; dyi < 2; ++dyi) {
      int   yy = y0 + dyi;
      float wy = dyi ? ly : 1.f - ly;
      if (yy < 0 || yy >= HH) continue;
#pragma unroll
      for (int dxi = 0; dxi < 2; ++dxi) {
        int   xx = x0 + dxi;
        float wx = dxi ? lx : 1.f - lx;
        if (xx < 0 || xx >= WW) continue;
        v += xc[yy * WW + xx] * (wy * wx);
      }
    }
    int k = kbase + kk;
    S[(size_t)(k >> 5) * ((size_t)NN * 32) + n32 + (k & 31)] = (_Float16)v;
  }
}

// ---------------------------------------------------------------------------
// 4) WMMA GEMM: out[o][n] = sum_k W[o][k]*S[k][n] + b_def[o]
//    M=256, K=2304, N=16384. Block tile 128x128, 8 wave32s as 2(M)x4(N),
//    each wave = 64x32 tile = 4x2 v_wmma_f32_16x16x32_f16 fragments.
//    Double-buffered LDS fed by global_load_async_to_lds_b128.
// ---------------------------------------------------------------------------
__global__ __launch_bounds__(256)
__attribute__((amdgpu_waves_per_eu(2)))
void k_gemm(const _Float16* __restrict__ Whk,
            const _Float16* __restrict__ S,
            const float* __restrict__ bdef,
            float* __restrict__ out) {
  __shared__ _Float16 As[2][128 * LDROW];   // 128 rows x 32k (+pad), x2 buffers
  __shared__ _Float16 Bs[2][128 * LDROW];   // 128 cols x 32k (+pad), x2 buffers

  const int t     = threadIdx.x;
  const int lane  = t & 31;
  const int wid   = t >> 5;        // 0..7
  const int waveM = wid & 1;       // 2 waves along M (64 rows each)
  const int waveN = wid >> 1;      // 4 waves along N (32 cols each)
  const int m0blk = blockIdx.x * 128;
  const int n0blk = blockIdx.y * 128;

  const int hsel = (lane >> 4) & 1;
  const int l15  = lane & 15;

  // staging chunk assignments (16B chunks; regions are fully contiguous)
  // A tile: 128 rows * 32 halves = 512 chunks -> thread handles t, t+256
  // B tile: 128 cols * 32 halves = 512 chunks -> thread handles t, t+256
  const int c0row = t >> 2,         c0sub = t & 3;           // chunk t
  const int c1row = (t + 256) >> 2, c1sub = (t + 256) & 3;   // chunk t+256

  const uint32_t ldsA0[2] = {
    (uint32_t)(uintptr_t)&As[0][c0row * LDROW + c0sub * 8],
    (uint32_t)(uintptr_t)&As[1][c0row * LDROW + c0sub * 8] };
  const uint32_t ldsA1[2] = {
    (uint32_t)(uintptr_t)&As[0][c1row * LDROW + c1sub * 8],
    (uint32_t)(uintptr_t)&As[1][c1row * LDROW + c1sub * 8] };
  const uint32_t ldsB0[2] = {
    (uint32_t)(uintptr_t)&Bs[0][c0row * LDROW + c0sub * 8],
    (uint32_t)(uintptr_t)&Bs[1][c0row * LDROW + c0sub * 8] };
  const uint32_t ldsB1[2] = {
    (uint32_t)(uintptr_t)&Bs[1][c1row * LDROW + c1sub * 8],
    (uint32_t)(uintptr_t)&Bs[0][c1row * LDROW + c1sub * 8] };
  // note: ldsB1 order swapped; indexed below with cur (not nxt) to reach Bs[nxt]

  // global chunk pointers (Whk: [kb][256][32], S: [kb][16384][32])
  const _Float16* gA0 = Whk + (size_t)(m0blk + c0row) * 32 + c0sub * 8;
  const _Float16* gA1 = Whk + (size_t)(m0blk + c1row) * 32 + c1sub * 8;
  const _Float16* gB0 = S   + (size_t)(n0blk + c0row) * 32 + c0sub * 8;
  const _Float16* gB1 = S   + (size_t)(n0blk + c1row) * 32 + c1sub * 8;
  const size_t strideA = (size_t)OO * 32;   // halves per k-tile of Whk
  const size_t strideB = (size_t)NN * 32;   // halves per k-tile of S

  F32x8 acc[4][2];
#pragma unroll
  for (int mi = 0; mi < 4; ++mi)
#pragma unroll
    for (int ni = 0; ni < 2; ++ni)
#pragma unroll
      for (int r = 0; r < 8; ++r) acc[mi][ni].f[r] = 0.f;

  // prologue: stage tile 0 into buffer 0
  async_b128(ldsA0[0], gA0);
  async_b128(ldsA1[0], gA1);
  async_b128(ldsB0[0], gB0);
  async_b128(ldsB1[1], gB1);   // ldsB1[1] holds &Bs[0][chunk t+256]
  wait_async0();
  __syncthreads();

  for (int it = 0; it < NKT; ++it) {
    const int cur = it & 1;
    const int nxt = cur ^ 1;

    // kick async staging of next tile (prev readers of buf nxt fenced by the
    // barrier at the end of iteration it-1)
    if (it + 1 < NKT) {
      const size_t ko = (size_t)(it + 1);
      async_b128(ldsA0[nxt], gA0 + ko * strideA);
      async_b128(ldsA1[nxt], gA1 + ko * strideA);
      async_b128(ldsB0[nxt], gB0 + ko * strideB);
      async_b128(ldsB1[cur], gB1 + ko * strideB);  // ldsB1[cur] == &Bs[nxt][...]
    }

    // --- B fragments: 2x aligned b128 per fragment ---
    F16x16 bfr[2];
#pragma unroll
    for (int ni = 0; ni < 2; ++ni) {
      const _Float16* pb = &Bs[cur][(waveN * 32 + ni * 16 + l15) * LDROW + hsel * 8];
      *(uint4*)&bfr[ni].h[0] = *(const uint4*)pb;        // K kb..kb+7
      *(uint4*)&bfr[ni].h[8] = *(const uint4*)(pb + 16); // K 16+kb..16+kb+7
    }

    // --- A fragments + WMMA (4x2 = 8 WMMAs per iteration) ---
#pragma unroll
    for (int mi = 0; mi < 4; ++mi) {
      F16x16 afr;
      const _Float16* pa = &As[cur][(waveM * 64 + mi * 16 + l15) * LDROW + hsel * 8];
      *(uint4*)&afr.h[0] = *(const uint4*)pa;            // K kb..kb+7
      *(uint4*)&afr.h[8] = *(const uint4*)(pa + 16);     // K 16+kb..16+kb+7
#pragma unroll
      for (int ni = 0; ni < 2; ++ni) {
        acc[mi][ni].v = __builtin_amdgcn_wmma_f32_16x16x32_f16(
            false, afr.v, false, bfr[ni].v,
            (short)0, acc[mi][ni].v, false, false);
      }
    }

    if (it + 1 < NKT) wait_async0();   // this wave's async stores landed in LDS
    __syncthreads();                   // all waves' stores landed / reads done
  }

  // --- epilogue: C layout VGPR r <-> row r + 8*(lane>=16), col = lane&15 ---
#pragma unroll
  for (int mi = 0; mi < 4; ++mi)
#pragma unroll
    for (int ni = 0; ni < 2; ++ni) {
      int nb = n0blk + waveN * 32 + ni * 16 + l15;
      int b  = nb >> 12;            // 128 | 4096 -> tile never straddles batches
      int hw = nb & (HW - 1);
#pragma unroll
      for (int r = 0; r < 8; ++r) {
        int row = m0blk + waveM * 64 + mi * 16 + hsel * 8 + r;
        out[((size_t)(b * OO + row) << 12) + hw] = acc[mi][ni].f[r] + bdef[row];
      }
    }
}

// ---------------------------------------------------------------------------
// launcher
// ---------------------------------------------------------------------------
extern "C" void kernel_launch(void* const* d_in, const int* in_sizes, int n_in,
                              void* d_out, int out_size, void* d_ws, size_t ws_size,
                              hipStream_t stream) {
  const float* x     = (const float*)d_in[0];   // (4,256,64,64)
  const float* w_off = (const float*)d_in[1];   // (18,256,3,3)
  const float* b_off = (const float*)d_in[2];   // (18,)
  const float* w_def = (const float*)d_in[3];   // (256,256,3,3)
  const float* b_def = (const float*)d_in[4];   // (256,)
  float* out = (float*)d_out;                   // (4,256,64,64)

  const size_t WH_BYTES = (size_t)OO * CK * sizeof(_Float16);   // 1,179,648
  const size_t S_BYTES  = (size_t)CK * NN * sizeof(_Float16);   // 75,497,472
  char* ws = (char*)d_ws;
  _Float16* Whk  = (_Float16*)ws;
  _Float16* S    = (_Float16*)(ws + WH_BYTES);
  float*    offs = (float*)(ws + WH_BYTES + S_BYTES);           // 1,179,648 B

  (void)in_sizes; (void)n_in; (void)out_size; (void)ws_size;

  k_wcvt<<<dim3(OO * CK / 256), dim3(256), 0, stream>>>(w_def, Whk);
  k_offconv<<<dim3(BB * 18 * HW / 256), dim3(256), 0, stream>>>(x, w_off, b_off, offs);
  k_sample<<<dim3(BB * CC * HW / 256), dim3(256), 0, stream>>>(x, offs, S);
  k_gemm<<<dim3(OO / 128, NN / 128), dim3(256), 0, stream>>>(Whk, S, b_def, out);
}
